// MultiHeadedAttention_47691316854980
// MI455X (gfx1250) — compile-verified, matrix operand order checked by reading, not run on hardware
//
#include <hip/hip_runtime.h>
#include <hip/hip_bf16.h>

typedef _Float16 half_t;
typedef _Float16 v16h __attribute__((ext_vector_type(16)));
typedef _Float16 v8h  __attribute__((ext_vector_type(8)));
typedef float    v8f  __attribute__((ext_vector_type(8)));
typedef float    v4f  __attribute__((ext_vector_type(4)));
typedef unsigned int v4u __attribute__((ext_vector_type(4)));
typedef int      v8i  __attribute__((ext_vector_type(8)));
typedef int      v4i  __attribute__((ext_vector_type(4)));

#define B_  2
#define S_  2048
#define D_  1024
#define H_  16
#define HD_ 64

// ---------------------------------------------------------------------------
// TDM: issue a 2D tensor-tile load (f32 elements) from global into LDS.
// D# packed per CDNA5 ISA.
// ---------------------------------------------------------------------------
__device__ __forceinline__
void tdm_load_tile_f32(const float* gptr, unsigned lds_off,
                       unsigned tile_d0, unsigned tile_d1,
                       unsigned tensor_d0, unsigned tensor_d1,
                       unsigned stride_d0) {
    unsigned long long ga = (unsigned long long)(uintptr_t)gptr;
    v4u g0;
    g0.x = 1u;                                            // count=1, user D#
    g0.y = lds_off;                                       // LDS byte address
    g0.z = (unsigned)ga;                                  // global_addr[31:0]
    g0.w = ((unsigned)(ga >> 32) & 0x01FFFFFFu)           // global_addr[56:32]
         | (2u << 30);                                    // type=2 ("image")
    v8i g1;
    g1[0] = (int)(2u << 16);                              // data_size=2 (4 bytes)
    g1[1] = (int)(tensor_d0 << 16);                       // tensor_dim0[15:0]
    g1[2] = (int)((tensor_d0 >> 16) | (tensor_d1 << 16));
    g1[3] = (int)((tensor_d1 >> 16) | (tile_d0 << 16));
    g1[4] = (int)(tile_d1 & 0xFFFFu);                     // tile_dim1, tile_dim2=0
    g1[5] = (int)stride_d0;                               // tensor_dim0_stride lo32
    g1[6] = 0;
    g1[7] = 0;
    v4i z4 = {0, 0, 0, 0};
#if __clang_major__ >= 23
    v8i z8 = {0, 0, 0, 0, 0, 0, 0, 0};
    __builtin_amdgcn_tensor_load_to_lds(g0, g1, z4, z4, z8, 0);
#else
    __builtin_amdgcn_tensor_load_to_lds(g0, g1, z4, z4, 0);
#endif
}

// ---------------------------------------------------------------------------
// Projection GEMM: Y[m,n] = sum_d X[m,d] * W[n,d] + bias[n]
// Double-buffered TDM staging: wave0 pre-issues tile k+1 into the alternate
// LDS buffer, waits tensorcnt<=2 (previous pair landed, new pair in flight),
// so the DMA overlaps the WMMA work of all 8 waves. Optional fused RoPE;
// stores f16 to [B,H,S,HD]. Block covers 32 rows x 64 cols.
// ---------------------------------------------------------------------------
__global__ __launch_bounds__(256)
void proj_rope_kernel(const float* __restrict__ X, const float* __restrict__ W,
                      const float* __restrict__ bias, half_t* __restrict__ out,
                      int do_rope) {
    __shared__ float Xtile[2][32][32];   // [buf][row within block][k]
    __shared__ float Wtile[2][64][32];   // [buf][col within block][k]

    const int lane = threadIdx.x & 31;
    const int wave = threadIdx.x >> 5;
    const int mrow = (wave >> 2) * 16;
    const int ncol = (wave & 3) * 16;
    const int row0 = blockIdx.x * 32 + mrow;
    const int col0 = blockIdx.y * 64 + ncol;

    const int r      = lane & 15;
    const int hiK    = (lane < 16) ? 0 : 8;
    const int rowOff = (lane < 16) ? 0 : 8;

    const float* Xblk = X + (size_t)blockIdx.x * 32 * D_;
    const float* Wblk = W + (size_t)blockIdx.y * 64 * D_;
    const unsigned ldsX0 = (unsigned)(uintptr_t)&Xtile[0][0][0];
    const unsigned ldsX1 = (unsigned)(uintptr_t)&Xtile[1][0][0];
    const unsigned ldsW0 = (unsigned)(uintptr_t)&Wtile[0][0][0];
    const unsigned ldsW1 = (unsigned)(uintptr_t)&Wtile[1][0][0];

    const int nIter = D_ / 32;

    if (wave == 0) {
        tdm_load_tile_f32(Xblk, ldsX0, 32, 32, D_, B_ * S_, D_);
        tdm_load_tile_f32(Wblk, ldsW0, 32, 64, D_, D_,      D_);
    }

    v8f c = {};
    for (int it = 0; it < nIter; ++it) {
        const int buf = it & 1;
        if (wave == 0) {
            if (it + 1 < nIter) {
                const int k1 = (it + 1) * 32;
                tdm_load_tile_f32(Xblk + k1, buf ? ldsX0 : ldsX1, 32, 32, D_, B_ * S_, D_);
                tdm_load_tile_f32(Wblk + k1, buf ? ldsW0 : ldsW1, 32, 64, D_, D_,      D_);
                __builtin_amdgcn_s_wait_tensorcnt(2);  // current pair landed
            } else {
                __builtin_amdgcn_s_wait_tensorcnt(0);  // final pair landed
            }
        }
        __syncthreads();          // current buffer visible to all waves

        // A fragment: 16x32 f16 from LDS X tile
        const float* xr = &Xtile[buf][mrow + r][0];
        v4f f0 = *(const v4f*)(xr + hiK);
        v4f f1 = *(const v4f*)(xr + hiK + 4);
        v4f f2 = *(const v4f*)(xr + hiK + 16);
        v4f f3 = *(const v4f*)(xr + hiK + 20);
        v16h a;
        #pragma unroll
        for (int i = 0; i < 4; ++i) {
            a[i]      = (half_t)f0[i];
            a[4 + i]  = (half_t)f1[i];
            a[8 + i]  = (half_t)f2[i];
            a[12 + i] = (half_t)f3[i];
        }
        // B fragment: 32x16 f16 from LDS W tile
        const float* wr = &Wtile[buf][ncol + r][0];
        v4f g0 = *(const v4f*)(wr + hiK);
        v4f g1 = *(const v4f*)(wr + hiK + 4);
        v4f g2 = *(const v4f*)(wr + hiK + 16);
        v4f g3 = *(const v4f*)(wr + hiK + 20);
        v16h bm;
        #pragma unroll
        for (int i = 0; i < 4; ++i) {
            bm[i]      = (half_t)g0[i];
            bm[4 + i]  = (half_t)g1[i];
            bm[8 + i]  = (half_t)g2[i];
            bm[12 + i] = (half_t)g3[i];
        }
        c = __builtin_amdgcn_wmma_f32_16x16x32_f16(false, a, false, bm,
                                                   (short)0, c, false, false);
        __syncthreads();          // all reads of this buffer done before re-issue
    }

    const int n_g = col0 + r;
    const float bb = bias[n_g];
    const int hd = n_g & (HD_ - 1);
    const int h  = n_g >> 6;

    float vals[8];
    #pragma unroll
    for (int i = 0; i < 8; ++i) vals[i] = c[i] + bb;

    if (do_rope) {
        const int p = hd >> 1;
        const float inv_freq = __expf(-(float)(2 * p) * (9.210340372f / (float)HD_));
        const bool even = (hd & 1) == 0;
        #pragma unroll
        for (int i = 0; i < 8; ++i) {
            int m = row0 + i + rowOff;
            int s = m & (S_ - 1);
            float ang = (float)s * inv_freq;
            float sn, cs;
            __sincosf(ang, &sn, &cs);
            float partner = __shfl_xor(vals[i], 1, 32);
            vals[i] = even ? (vals[i] * cs - partner * sn)
                           : (vals[i] * cs + partner * sn);
        }
    }

    #pragma unroll
    for (int i = 0; i < 8; ++i) {
        int m = row0 + i + rowOff;
        int b = m >> 11;
        int s = m & (S_ - 1);
        size_t idx = (((size_t)b * H_ + h) * S_ + s) * HD_ + hd;
        out[idx] = (half_t)vals[i];
    }
}

// ---------------------------------------------------------------------------
// Flash attention: one wave per 16-row Q tile per (b,h). Streams 32-key blocks.
// ---------------------------------------------------------------------------
__global__ __launch_bounds__(32)
void flash_attn_kernel(const half_t* __restrict__ Qh, const half_t* __restrict__ Kh,
                       const half_t* __restrict__ Vh, const int* __restrict__ mask,
                       half_t* __restrict__ xout) {
    __shared__ half_t pshare[16 * 32];

    const int lane = threadIdx.x & 31;
    const int q0 = blockIdx.x * 16;
    const int h  = blockIdx.y;
    const int b  = blockIdx.z;
    const size_t headBase = ((size_t)b * H_ + h) * (size_t)S_ * HD_;

    const int r      = lane & 15;
    const int hiK    = (lane < 16) ? 0 : 8;
    const int rowOff = (lane < 16) ? 0 : 8;

    v16h aq[2];
    #pragma unroll
    for (int t = 0; t < 2; ++t) {
        const half_t* qp = Qh + headBase + (size_t)(q0 + r) * HD_ + t * 32 + hiK;
        v8h q0v = *(const v8h*)(qp);
        v8h q1v = *(const v8h*)(qp + 16);
        #pragma unroll
        for (int i = 0; i < 8; ++i) {
            aq[t][i]     = (half_t)((float)q0v[i] * 0.125f);
            aq[t][8 + i] = (half_t)((float)q1v[i] * 0.125f);
        }
    }

    float mrun[8], lrun[8];
    #pragma unroll
    for (int i = 0; i < 8; ++i) { mrun[i] = -3.0e38f; lrun[i] = 0.0f; }
    v8f o[4] = {};

    for (int kb = 0; kb < S_; kb += 32) {
        if (kb + 32 < S_) {
            __builtin_prefetch(Kh + headBase + (size_t)(kb + 32 + r) * HD_, 0, 1);
            __builtin_prefetch(Vh + headBase + (size_t)(kb + 32 + r) * HD_, 0, 1);
        }
        v8f sc[2];
        #pragma unroll
        for (int t = 0; t < 2; ++t) {
            v8f acc = {};
            #pragma unroll
            for (int kk = 0; kk < 2; ++kk) {
                const half_t* kp = Kh + headBase +
                                   (size_t)(kb + t * 16 + r) * HD_ + kk * 32 + hiK;
                v8h k0v = *(const v8h*)(kp);
                v8h k1v = *(const v8h*)(kp + 16);
                v16h bk;
                #pragma unroll
                for (int i = 0; i < 8; ++i) { bk[i] = k0v[i]; bk[8 + i] = k1v[i]; }
                acc = __builtin_amdgcn_wmma_f32_16x16x32_f16(false, aq[kk], false, bk,
                                                             (short)0, acc, false, false);
            }
            sc[t] = acc;
        }
        #pragma unroll
        for (int t = 0; t < 2; ++t) {
            #pragma unroll
            for (int i = 0; i < 8; ++i) {
                int qg = q0 + i + rowOff;
                int kg = kb + t * 16 + r;
                int mk = mask[((size_t)b * S_ + qg) * S_ + kg];
                if (mk == 0) sc[t][i] = -1.0e9f;
            }
        }
        float p0[8], p1[8];
        #pragma unroll
        for (int i = 0; i < 8; ++i) {
            float mb = fmaxf(sc[0][i], sc[1][i]);
            #pragma unroll
            for (int xm = 1; xm < 16; xm <<= 1) mb = fmaxf(mb, __shfl_xor(mb, xm, 32));
            float mnew  = fmaxf(mrun[i], mb);
            float alpha = __expf(mrun[i] - mnew);
            p0[i] = __expf(sc[0][i] - mnew);
            p1[i] = __expf(sc[1][i] - mnew);
            float ls = p0[i] + p1[i];
            #pragma unroll
            for (int xm = 1; xm < 16; xm <<= 1) ls += __shfl_xor(ls, xm, 32);
            lrun[i] = lrun[i] * alpha + ls;
            mrun[i] = mnew;
            #pragma unroll
            for (int j = 0; j < 4; ++j) o[j][i] *= alpha;
        }
        #pragma unroll
        for (int i = 0; i < 8; ++i) {
            pshare[(i + rowOff) * 32 + r]      = (half_t)p0[i];
            pshare[(i + rowOff) * 32 + 16 + r] = (half_t)p1[i];
        }
        v16h pa;
        #pragma unroll
        for (int i = 0; i < 8; ++i) {
            pa[i]     = pshare[r * 32 + hiK + i];
            pa[8 + i] = pshare[r * 32 + hiK + 16 + i];
        }
        #pragma unroll
        for (int j = 0; j < 4; ++j) {
            v16h bv;
            #pragma unroll
            for (int i = 0; i < 8; ++i) {
                bv[i]     = Vh[headBase + (size_t)(kb + hiK + i) * HD_ + j * 16 + r];
                bv[8 + i] = Vh[headBase + (size_t)(kb + hiK + 16 + i) * HD_ + j * 16 + r];
            }
            o[j] = __builtin_amdgcn_wmma_f32_16x16x32_f16(false, pa, false, bv,
                                                          (short)0, o[j], false, false);
        }
    }

    #pragma unroll
    for (int i = 0; i < 8; ++i) {
        float inv_l = 1.0f / lrun[i];
        int qg = q0 + i + rowOff;
        #pragma unroll
        for (int j = 0; j < 4; ++j) {
            size_t idx = ((size_t)b * S_ + qg) * D_ + h * HD_ + j * 16 + r;
            xout[idx] = (half_t)(o[j][i] * inv_l);
        }
    }
}

// ---------------------------------------------------------------------------
// Output projection: out[m,n] = sum_d Xh[m,d] * Wo[n,d] + bo[n], f32 output.
// ---------------------------------------------------------------------------
__global__ __launch_bounds__(256)
void out_proj_kernel(const half_t* __restrict__ Xh, const float* __restrict__ W,
                     const float* __restrict__ bias, float* __restrict__ out) {
    const int lane = threadIdx.x & 31;
    const int wave = threadIdx.x >> 5;
    const int row0 = blockIdx.x * 32 + (wave >> 2) * 16;
    const int col0 = blockIdx.y * 64 + (wave & 3) * 16;

    const int r      = lane & 15;
    const int hiK    = (lane < 16) ? 0 : 8;
    const int rowOff = (lane < 16) ? 0 : 8;

    const half_t* xrow = Xh + (size_t)(row0 + r) * D_;
    const float*  wrow = W  + (size_t)(col0 + r) * D_;

    v8f c = {};
    for (int k0 = 0; k0 < D_; k0 += 32) {
        if (k0 + 128 < D_) {
            __builtin_prefetch(xrow + k0 + 128, 0, 1);
            __builtin_prefetch(wrow + k0 + 128, 0, 1);
        }
        v8h x0 = *(const v8h*)(xrow + k0 + hiK);
        v8h x1 = *(const v8h*)(xrow + k0 + hiK + 16);
        v16h a;
        #pragma unroll
        for (int i = 0; i < 8; ++i) { a[i] = x0[i]; a[8 + i] = x1[i]; }

        const v4f* wa = (const v4f*)(wrow + k0 + hiK);
        v4f g0 = wa[0], g1 = wa[1];
        const v4f* wb = (const v4f*)(wrow + k0 + hiK + 16);
        v4f g2 = wb[0], g3 = wb[1];
        v16h bm;
        #pragma unroll
        for (int i = 0; i < 4; ++i) {
            bm[i]      = (half_t)g0[i];
            bm[4 + i]  = (half_t)g1[i];
            bm[8 + i]  = (half_t)g2[i];
            bm[12 + i] = (half_t)g3[i];
        }
        c = __builtin_amdgcn_wmma_f32_16x16x32_f16(false, a, false, bm,
                                                   (short)0, c, false, false);
    }

    const float bb = bias[col0 + r];
    #pragma unroll
    for (int i = 0; i < 8; ++i) {
        out[(size_t)(row0 + i + rowOff) * D_ + col0 + r] = c[i] + bb;
    }
}

// ---------------------------------------------------------------------------
extern "C" void kernel_launch(void* const* d_in, const int* in_sizes, int n_in,
                              void* d_out, int out_size, void* d_ws, size_t ws_size,
                              hipStream_t stream) {
    (void)in_sizes; (void)n_in; (void)out_size; (void)ws_size;
    const float* q    = (const float*)d_in[0];
    const float* k    = (const float*)d_in[1];
    const float* v    = (const float*)d_in[2];
    const float* Wq   = (const float*)d_in[3];
    const float* bq   = (const float*)d_in[4];
    const float* Wk   = (const float*)d_in[5];
    const float* bk   = (const float*)d_in[6];
    const float* Wv   = (const float*)d_in[7];
    const float* bv   = (const float*)d_in[8];
    const float* Wo   = (const float*)d_in[9];
    const float* bo   = (const float*)d_in[10];
    const int*   mask = (const int*)d_in[11];
    float* out = (float*)d_out;

    const size_t elems = (size_t)B_ * H_ * S_ * HD_;
    half_t* Qh = (half_t*)d_ws;
    half_t* Kh = Qh + elems;
    half_t* Vh = Kh + elems;
    half_t* Xh = Vh + elems;

    dim3 gBlock(256);
    dim3 gGrid((B_ * S_) / 32, D_ / 64);

    proj_rope_kernel<<<gGrid, gBlock, 0, stream>>>(q, Wq, bq, Qh, 1);
    proj_rope_kernel<<<gGrid, gBlock, 0, stream>>>(k, Wk, bk, Kh, 1);
    proj_rope_kernel<<<gGrid, gBlock, 0, stream>>>(v, Wv, bv, Vh, 0);

    flash_attn_kernel<<<dim3(S_ / 16, H_, B_), dim3(32), 0, stream>>>(Qh, Kh, Vh, mask, Xh);

    out_proj_kernel<<<gGrid, gBlock, 0, stream>>>(Xh, Wo, bo, out);
}
